// MultiHeadGraphAttentionModule_11759620456488
// MI455X (gfx1250) — matrix operand order repaired and checked by reading, not compiled, not run on hardware
//
#include <hip/hip_runtime.h>
#include <hip/hip_bf16.h>

typedef __attribute__((ext_vector_type(16))) _Float16     v16h;
typedef __attribute__((ext_vector_type(8)))  float        v8f;
typedef __attribute__((ext_vector_type(4)))  unsigned int u32x4;
typedef __attribute__((ext_vector_type(8)))  int          i32x8;
typedef __attribute__((ext_vector_type(4)))  int          i32x4;

#define N_NODES 8192
#define F_IN    128
#define F_OUT   32
#define NHEAD   4
#define BJ      256   // j-tile DMA'd into LDS per block (16 x 256 ints = 16 KB + pad)
#define NJT     (N_NODES / 32)

// ---------------------------------------------------------------------------
// TDM: DMA one 16 x BJ int32 tile of adj into LDS, with 4-DWORD row padding
// (pad_interval=7 -> every 256 DWORDs, pad_amount=3 -> 4 DWORDs) so LDS rows
// land on distinct banks.  Tracked by TENSORcnt.
// ---------------------------------------------------------------------------
__device__ __forceinline__ void tdm_load_adj_tile(void* ldsDst, const int* gsrc) {
    unsigned int       lds = (unsigned int)(unsigned long long)ldsDst; // low32 = LDS offset
    unsigned long long ga  = (unsigned long long)gsrc;
    u32x4 g0 = {};
    g0[0] = 1u;                                                  // count=1 valid descriptor
    g0[1] = lds;                                                 // lds_addr
    g0[2] = (unsigned int)(ga & 0xFFFFFFFFull);                  // global_addr[31:0]
    g0[3] = (unsigned int)((ga >> 32) & 0x1FFFFFFull) | (2u << 30); // addr[56:32] | type=2
    i32x8 g1 = {};
    g1[0] = (int)((2u << 16) | (1u << 20) | (7u << 22) | (3u << 25)); // 4B elems, pad en, 256dw/4dw
    g1[1] = (int)(((unsigned)N_NODES & 0xFFFFu) << 16);          // tensor_dim0 lo16
    g1[2] = (int)((((unsigned)N_NODES) >> 16) | (((unsigned)N_NODES & 0xFFFFu) << 16)); // d0 hi | d1 lo
    g1[3] = (int)((((unsigned)N_NODES) >> 16) | ((unsigned)BJ << 16)); // d1 hi | tile_dim0=BJ
    g1[4] = 16;                                                  // tile_dim1=16 rows, tile_dim2=0
    g1[5] = N_NODES;                                             // tensor_dim0_stride lo32
    g1[6] = 0;
    g1[7] = 0;
    i32x4 z4 = {};
#if __clang_major__ >= 23
    i32x8 z8 = {};
    __builtin_amdgcn_tensor_load_to_lds(g0, g1, z4, z4, z8, 0);
#else
    __builtin_amdgcn_tensor_load_to_lds(g0, g1, z4, z4, 0);
#endif
}

// ---------------------------------------------------------------------------
// Kernel 1: projection h[h,n,:] = X[n,:] @ W[h,:,:]  via WMMA f16 -> f32 acc.
// ---------------------------------------------------------------------------
__global__ __launch_bounds__(128) void proj_kernel(const float* __restrict__ X,
                                                   const float* __restrict__ W,
                                                   _Float16* __restrict__ hbuf) {
    __shared__ _Float16 xt[16][F_IN + 8];
    const int i0  = blockIdx.x * 16;
    const int tid = threadIdx.x;

    for (int idx = tid; idx < 16 * F_IN; idx += 128) {
        int r = idx >> 7, c = idx & 127;
        xt[r][c] = (_Float16)X[(size_t)(i0 + r) * F_IN + c];
    }
    __syncthreads();

    const int lane = tid & 31;
    const int head = tid >> 5;
    const int lrow = lane & 15;
    const int lhi  = lane >> 4;

    v8f acc0 = {}, acc1 = {};
    const float* Wg = W + (size_t)head * F_IN * F_OUT;

    for (int k0 = 0; k0 < F_IN; k0 += 32) {
        union { v16h v; _Float16 e[16]; } a, b0, b1;
#pragma unroll
        for (int v = 0; v < 8; ++v) {
            int kra = (v & 3) * 2 + (lhi << 3) + ((v >> 2) << 4);  // A 16x32 f16 layout
            a.e[2 * v]     = xt[lrow][k0 + kra];
            a.e[2 * v + 1] = xt[lrow][k0 + kra + 1];
            int krb = (v & 3) * 2 + (lhi << 3) + ((v >> 2) << 4);  // B 32x16 f16 layout
            b0.e[2 * v]     = (_Float16)Wg[(k0 + krb) * F_OUT + lrow];
            b0.e[2 * v + 1] = (_Float16)Wg[(k0 + krb + 1) * F_OUT + lrow];
            b1.e[2 * v]     = (_Float16)Wg[(k0 + krb) * F_OUT + lrow + 16];
            b1.e[2 * v + 1] = (_Float16)Wg[(k0 + krb + 1) * F_OUT + lrow + 16];
        }
        acc0 = __builtin_amdgcn_wmma_f32_16x16x32_f16(false, a.v, false, b0.v, (short)0, acc0, false, false);
        acc1 = __builtin_amdgcn_wmma_f32_16x16x32_f16(false, a.v, false, b1.v, (short)0, acc1, false, false);
    }

    _Float16* hb = hbuf + (size_t)head * N_NODES * F_OUT;
#pragma unroll
    for (int r = 0; r < 8; ++r) {
        int m = r + (lhi << 3);
        hb[(size_t)(i0 + m) * F_OUT + lrow]      = (_Float16)acc0[r];
        hb[(size_t)(i0 + m) * F_OUT + lrow + 16] = (_Float16)acc1[r];
    }
}

// ---------------------------------------------------------------------------
// Kernel 2: s[h,n] = h . a_src ; d[h,n] = h . a_dst
// ---------------------------------------------------------------------------
__global__ __launch_bounds__(256) void sd_kernel(const _Float16* __restrict__ hbuf,
                                                 const float* __restrict__ a_src,
                                                 const float* __restrict__ a_dst,
                                                 float* __restrict__ s,
                                                 float* __restrict__ d) {
    int idx = blockIdx.x * 256 + threadIdx.x;
    if (idx >= NHEAD * N_NODES) return;
    int head = idx / N_NODES;
    const _Float16* hr = hbuf + (size_t)idx * F_OUT;
    const float* as = a_src + head * F_OUT;
    const float* ad = a_dst + head * F_OUT;
    float ss = 0.f, dd = 0.f;
#pragma unroll
    for (int o = 0; o < F_OUT; ++o) {
        float hv = (float)hr[o];
        ss += hv * as[o];
        dd += hv * ad[o];
    }
    s[idx] = ss;
    d[idx] = dd;
}

// ---------------------------------------------------------------------------
// Kernel 3: per-head max of d[]
// ---------------------------------------------------------------------------
__global__ __launch_bounds__(256) void maxd_kernel(const float* __restrict__ d,
                                                   float* __restrict__ maxd) {
    __shared__ float red[256];
    const float* dh = d + (size_t)blockIdx.x * N_NODES;
    float m = -1e30f;
    for (int n = threadIdx.x; n < N_NODES; n += 256) m = fmaxf(m, dh[n]);
    red[threadIdx.x] = m;
    __syncthreads();
    for (int st = 128; st > 0; st >>= 1) {
        if (threadIdx.x < st) red[threadIdx.x] = fmaxf(red[threadIdx.x], red[threadIdx.x + st]);
        __syncthreads();
    }
    if (threadIdx.x == 0) maxd[blockIdx.x] = red[0];
}

// ---------------------------------------------------------------------------
// Kernel 4: factored-softmax coefficients (see round-0 derivation; all <= 1).
// ---------------------------------------------------------------------------
__global__ __launch_bounds__(256) void prep_kernel(const float* __restrict__ s,
                                                   const float* __restrict__ d,
                                                   const float* __restrict__ maxd,
                                                   float4* __restrict__ RS,
                                                   float4* __restrict__ CS) {
    int idx = blockIdx.x * 256 + threadIdx.x;
    if (idx >= NHEAD * N_NODES) return;
    int head = idx / N_NODES;
    float md = maxd[head];
    float sv = s[idx], dv = d[idx];
    float u = sv + md;
    float m = fmaxf(u, 0.2f * u);
    RS[idx] = make_float4(__expf(u - m), __expf(0.2f * u - m), -sv, 0.f);
    CS[idx] = make_float4(__expf(dv - md), __expf(0.2f * (dv - md)), dv, 0.f);
}

// ---------------------------------------------------------------------------
// Kernel 5: pre-gather CS into the per-(jtile, lane-half) A-fragment j order,
// so the hot loop reads 256 contiguous bytes per lane (b128 clauses).
// ---------------------------------------------------------------------------
__global__ __launch_bounds__(256) void cswiz_kernel(const float4* __restrict__ CS,
                                                    float4* __restrict__ csw) {
    int idx = blockIdx.x * 256 + threadIdx.x;   // [head][jt][half]
    if (idx >= NHEAD * NJT * 2) return;
    int half = idx & 1;
    int jt   = (idx >> 1) & (NJT - 1);
    int head = idx >> 1 >> 8;
    const float4* cs  = CS + (size_t)head * N_NODES + jt * 32;
    float4*       dst = csw + (size_t)idx * 16;
#pragma unroll
    for (int v = 0; v < 8; ++v) {
        int jra = (v & 3) * 2 + (half << 3) + ((v >> 2) << 4);
        dst[2 * v]     = cs[jra];
        dst[2 * v + 1] = cs[jra + 1];
    }
}

// ---------------------------------------------------------------------------
// Kernel 6: repack h into the exact B-fragment register image:
// hs[head][jt][ntile][lane][16 halves] -> one 32-byte v16h load per fragment.
// ---------------------------------------------------------------------------
__global__ __launch_bounds__(256) void repack_kernel(const _Float16* __restrict__ hbuf,
                                                     _Float16* __restrict__ hs) {
    int idx = blockIdx.x * 256 + threadIdx.x;   // [head][jt][ntile][lane]
    if (idx >= NHEAD * NJT * 2 * 32) return;
    int lane  = idx & 31;
    int ntile = (idx >> 5) & 1;
    int jt    = (idx >> 6) & (NJT - 1);
    int head  = idx >> 14;
    int n     = (lane & 15) + (ntile << 4);
    int khalf = lane >> 4;
    const _Float16* hb = hbuf + ((size_t)head * N_NODES + jt * 32) * F_OUT;
    union { v16h v; _Float16 e[16]; } t;
#pragma unroll
    for (int v = 0; v < 8; ++v) {
        int k = (v & 3) * 2 + (khalf << 3) + ((v >> 2) << 4);
        t.e[2 * v]     = hb[(size_t)k * F_OUT + n];
        t.e[2 * v + 1] = hb[(size_t)(k + 1) * F_OUT + n];
    }
    ((v16h*)hs)[idx] = t.v;
}

// ---------------------------------------------------------------------------
// Kernel 7: fused flash-GAT.  Block = 16 rows x 4 heads (adj read exactly once,
// DMA'd by the TDM with double buffering).  Row sums via a third WMMA vs ones.
// ---------------------------------------------------------------------------
__global__ __launch_bounds__(128) void gat_kernel(const int* __restrict__ adj,
                                                  const _Float16* __restrict__ hs,
                                                  const float4* __restrict__ RS,
                                                  const float4* __restrict__ csw,
                                                  float* __restrict__ out) {
    __shared__ int adjt[2][16][BJ + 4];   // TDM pads rows by 4 DWORDs -> bank-clean

    const int i0   = blockIdx.x * 16;
    const int tid  = threadIdx.x;
    const int wave = tid >> 5;
    const int lane = tid & 31;
    const int head = wave;
    const int lrow = lane & 15;
    const int lhi  = lane >> 4;

    const float4 rs = RS[(size_t)head * N_NODES + i0 + lrow];
    const float Ai = rs.x, Ci = rs.y, ti = rs.z;
    const v16h* hsv = (const v16h*)hs;

    union { v16h v; _Float16 e[16]; } ones;
#pragma unroll
    for (int t = 0; t < 16; ++t) ones.e[t] = (_Float16)1.0f;

    v8f acc0 = {}, acc1 = {}, accs = {};

    if (wave == 0) tdm_load_adj_tile(&adjt[0][0][0], adj + (size_t)i0 * N_NODES);
    int p = 0;
    for (int j0 = 0; j0 < N_NODES; j0 += BJ) {
        if (wave == 0) __builtin_amdgcn_s_wait_tensorcnt(0);
        __syncthreads();                        // tile p ready; tile p^1 free
        if (wave == 0 && j0 + BJ < N_NODES)
            tdm_load_adj_tile(&adjt[p ^ 1][0][0], adj + (size_t)i0 * N_NODES + j0 + BJ);

        for (int jj = 0; jj < BJ; jj += 32) {
            const int jt = (j0 + jj) >> 5;
            const float4* csl = csw + ((size_t)((head * NJT + jt) * 2 + lhi)) * 16;
            union { v16h v; _Float16 e[16]; } P;
#pragma unroll
            for (int v = 0; v < 8; ++v) {
                int jra = (v & 3) * 2 + (lhi << 3) + ((v >> 2) << 4);
#pragma unroll
                for (int e = 0; e < 2; ++e) {
                    float4 cs = csl[2 * v + e];                 // (B, D, d)
                    int    am = adjt[p][lrow][jj + jra + e];
                    float  w  = (cs.z >= ti) ? (Ai * cs.x) : (Ci * cs.y);
                    w         = (am > 0) ? w : 0.f;
                    P.e[2 * v + e] = (_Float16)w;
                }
            }
            size_t ib = ((size_t)(head * NJT + jt) * 2) * 32 + lane;
            v16h B0 = hsv[ib];
            v16h B1 = hsv[ib + 32];
            acc0 = __builtin_amdgcn_wmma_f32_16x16x32_f16(false, P.v, false, B0,     (short)0, acc0, false, false);
            acc1 = __builtin_amdgcn_wmma_f32_16x16x32_f16(false, P.v, false, B1,     (short)0, acc1, false, false);
            accs = __builtin_amdgcn_wmma_f32_16x16x32_f16(false, P.v, false, ones.v, (short)0, accs, false, false);
        }
        p ^= 1;
    }

    // acc layout: VGPR r -> row m=r+8*lhi; accs holds the same row's sum in every column.
#pragma unroll
    for (int r = 0; r < 8; ++r) {
        int m = r + (lhi << 3);
        float inv = 1.0f / fmaxf(accs[r], 1e-30f);
        float* orow = out + (size_t)(i0 + m) * (NHEAD * F_OUT) + head * F_OUT;
        orow[lrow]      = acc0[r] * inv;
        orow[lrow + 16] = acc1[r] * inv;
    }
}

// ---------------------------------------------------------------------------
extern "C" void kernel_launch(void* const* d_in, const int* in_sizes, int n_in,
                              void* d_out, int out_size, void* d_ws, size_t ws_size,
                              hipStream_t stream) {
    const float* X     = (const float*)d_in[0];   // [8192,128]
    const int*   adj   = (const int*)  d_in[1];   // [8192,8192]
    const float* W     = (const float*)d_in[2];   // [4,128,32]
    const float* a_src = (const float*)d_in[3];   // [4,32]
    const float* a_dst = (const float*)d_in[4];   // [4,32]
    float*       outp  = (float*)d_out;           // [8192,128]

    char* ws = (char*)d_ws;
    _Float16* hbuf = (_Float16*)(ws);                              // [0, 2M)
    float*    s    = (float*)  (ws + (2u << 20));                  // 128 KB
    float*    d    = (float*)  (ws + (2u << 20) + (128u << 10));   // 128 KB
    float*    maxd = (float*)  (ws + (2u << 20) + (256u << 10));   // tiny
    float4*   RS   = (float4*) (ws + (2u << 20) + (512u << 10));   // 512 KB
    float4*   CS   = (float4*) (ws + (3u << 20));                  // 512 KB
    float4*   csw  = (float4*) (ws + (3u << 20) + (512u << 10));   // 512 KB
    _Float16* hs   = (_Float16*)(ws + (4u << 20));                 // 2 MB

    proj_kernel  <<<N_NODES / 16, 128, 0, stream>>>(X, W, hbuf);
    sd_kernel    <<<(NHEAD * N_NODES) / 256, 256, 0, stream>>>(hbuf, a_src, a_dst, s, d);
    maxd_kernel  <<<NHEAD, 256, 0, stream>>>(d, maxd);
    prep_kernel  <<<(NHEAD * N_NODES) / 256, 256, 0, stream>>>(s, d, maxd, RS, CS);
    cswiz_kernel <<<(NHEAD * NJT * 2 + 255) / 256, 256, 0, stream>>>(CS, csw);
    repack_kernel<<<(NHEAD * NJT * 2 * 32) / 256, 256, 0, stream>>>(hbuf, hs);
    gat_kernel   <<<N_NODES / 16, 128, 0, stream>>>(adj, hs, RS, csw, outp);
}
